// DeepInteractLayer_Base_31714038513810
// MI455X (gfx1250) — matrix operand order repaired and checked by reading, not compiled, not run on hardware
//
#include <hip/hip_runtime.h>
#include <hip/hip_bf16.h>
#include <cstdint>
#include <cstddef>

typedef __attribute__((ext_vector_type(16))) _Float16 v16h;
typedef __attribute__((ext_vector_type(8)))  _Float16 v8h;
typedef __attribute__((ext_vector_type(8)))  float    v8f;
typedef __attribute__((ext_vector_type(4)))  unsigned u32x4;
typedef __attribute__((ext_vector_type(8)))  int      i32x8;
typedef __attribute__((ext_vector_type(4)))  int      i32x4;

union Frag16 { v16h v; struct { v8h lo, hi; } p; };

static constexpr int Bb = 8, Nn = 2048, Dd = 512;
static constexpr int Mrows = Bb * Nn;           // 16384

#ifndef __has_builtin
#define __has_builtin(x) 0
#endif
#if __has_builtin(__builtin_amdgcn_tensor_load_to_lds)
#define USE_TDM 1
#else
#define USE_TDM 0
#endif

// ---------------------------------------------------------------------------
// WMMA helpers (CDNA5 16x16x32 f16 -> f32)
// ---------------------------------------------------------------------------
__device__ __forceinline__ v8f wmma32(v16h a, v16h b, v8f c) {
  return __builtin_amdgcn_wmma_f32_16x16x32_f16(false, a, false, b, (short)0, c,
                                                false, false);
}

// A-matrix fragment (16x32, row-major source, stride ld halves).
__device__ __forceinline__ v16h load_a_frag(const _Float16* base, int ld) {
  const int lane = threadIdx.x & 31;
  const int row  = lane & 15;
  const int grp  = lane >> 4;
  const _Float16* p = base + (size_t)row * ld + grp * 8;
  Frag16 f;
  f.p.lo = *(const v8h*)(p);
  f.p.hi = *(const v8h*)(p + 16);
  return f.v;
}

// B-matrix fragment (32x16) from K-major storage Bt[n][k] (stride ld halves).
__device__ __forceinline__ v16h load_bt_frag(const _Float16* bt, int ld) {
  const int lane = threadIdx.x & 31;
  const int col  = lane & 15;
  const int grp  = lane >> 4;
  const _Float16* p = bt + (size_t)col * ld + grp * 8;
  Frag16 f;
  f.p.lo = *(const v8h*)(p);
  f.p.hi = *(const v8h*)(p + 16);
  return f.v;
}

#if USE_TDM
// ---------------------------------------------------------------------------
// TDM 2D tile load: global (row-major, rowStride halves) -> LDS, f16 elements.
// Tile = tileD0 (contiguous) x tileD1 rows. LDS padding: after every tile row
// (32 halves = 16 DWORDs, pad_interval code 3) insert 4 DWORDs (code 3), so
// the LDS row stride is 40 halves — matching the fragment loaders.
// ---------------------------------------------------------------------------
__device__ __forceinline__ void tdm_load_2d(unsigned ldsAddr, const void* gptr,
                                            unsigned tensorD0, unsigned tensorD1,
                                            unsigned tileD0, unsigned tileD1,
                                            unsigned rowStride) {
  const unsigned long long ga = (unsigned long long)gptr;
  u32x4 g0;
  g0[0] = 1u;                                        // count=1, user mode
  g0[1] = ldsAddr;                                   // lds_addr (bytes)
  g0[2] = (unsigned)(ga & 0xffffffffu);              // global_addr[31:0]
  g0[3] = (unsigned)((ga >> 32) & 0x01ffffffu)       // global_addr[56:32]
          | (2u << 30);                              // type = 2 ("image")
  i32x8 g1;
  g1[0] = (int)((1u << 16)                           // data_size = 1 (2 bytes)
                | (1u << 20)                         // pad_enable
                | (3u << 22)                         // pad_interval: 16 DWORDs
                | (3u << 25));                       // pad_amount:   4 DWORDs
  g1[1] = (int)((tensorD0 & 0xffffu) << 16);         // tensor_dim0[15:0]
  g1[2] = (int)(((tensorD0 >> 16) & 0xffffu)         // tensor_dim0[31:16]
                | ((tensorD1 & 0xffffu) << 16));     // tensor_dim1[15:0]
  g1[3] = (int)(((tensorD1 >> 16) & 0xffffu)         // tensor_dim1[31:16]
                | ((tileD0 & 0xffffu) << 16));       // tile_dim0
  g1[4] = (int)(tileD1 & 0xffffu);                   // tile_dim1 (tile_dim2=0)
  g1[5] = (int)rowStride;                            // tensor_dim0_stride[31:0]
  g1[6] = 0;
  g1[7] = 0;
  i32x4 z4 = {0, 0, 0, 0};
#if __clang_major__ >= 23
  i32x8 z8 = {0, 0, 0, 0, 0, 0, 0, 0};
  __builtin_amdgcn_tensor_load_to_lds(g0, g1, z4, z4, z8, 0);
#else
  __builtin_amdgcn_tensor_load_to_lds(g0, g1, z4, z4, 0);
#endif
}
#endif  // USE_TDM

// ---------------------------------------------------------------------------
// fp32 -> f16 convert (grid-stride)
// ---------------------------------------------------------------------------
__global__ void f32_to_f16_kernel(const float* __restrict__ in,
                                  _Float16* __restrict__ out, int n) {
  int i = blockIdx.x * blockDim.x + threadIdx.x;
  int stride = gridDim.x * blockDim.x;
  for (; i < n; i += stride) out[i] = (_Float16)in[i];
}

// ---------------------------------------------------------------------------
// f16 transpose: [B][N][D] -> [B][D][N]
// ---------------------------------------------------------------------------
__global__ __launch_bounds__(256)
void transpose_f16_kernel(const _Float16* __restrict__ in,
                          _Float16* __restrict__ out) {
  __shared__ _Float16 tile[32][33];
  const int b = blockIdx.z;
  const int mBase = blockIdx.x * 32;
  const int dBase = blockIdx.y * 32;
  const int tx = threadIdx.x & 31;
  const int ty = threadIdx.x >> 5;   // 0..7
  const _Float16* ib = in  + (size_t)b * Nn * Dd;
  _Float16*       ob = out + (size_t)b * Dd * Nn;
  #pragma unroll
  for (int i = 0; i < 4; ++i) {
    int m = mBase + ty + i * 8;
    tile[ty + i * 8][tx] = ib[(size_t)m * Dd + dBase + tx];
  }
  __syncthreads();
  #pragma unroll
  for (int i = 0; i < 4; ++i) {
    int d = dBase + ty + i * 8;
    ob[(size_t)d * Nn + mBase + tx] = tile[tx][ty + i * 8];
  }
}

// ---------------------------------------------------------------------------
// Weight convert + transpose: W f32 [K][N] -> Wt f16 [N][K]  (K = N = Dd)
// ---------------------------------------------------------------------------
__global__ __launch_bounds__(256)
void convert_transpose_w_kernel(const float* __restrict__ in,
                                _Float16* __restrict__ out) {
  __shared__ float tile[32][33];
  const int kBase = blockIdx.x * 32;
  const int nBase = blockIdx.y * 32;
  const int tx = threadIdx.x & 31;
  const int ty = threadIdx.x >> 5;
  #pragma unroll
  for (int i = 0; i < 4; ++i)
    tile[ty + i * 8][tx] = in[(size_t)(kBase + ty + i * 8) * Dd + nBase + tx];
  __syncthreads();
  #pragma unroll
  for (int i = 0; i < 4; ++i)
    out[(size_t)(nBase + ty + i * 8) * Dd + kBase + tx] =
        (_Float16)tile[tx][ty + i * 8];
}

// ---------------------------------------------------------------------------
// Generic WMMA GEMM: out = epilogue(A[f16 MxK] * Wt[f16 NxK]^T + bias)
// Block: 128 threads = 4 waves (2x2), block tile 64x64, K step 32.
// Tiles staged in LDS by the Tensor Data Mover (double-buffered).
// MODE 0: f16 out.  MODE 1: f16 out, ELU.  MODE 2: f32 out.
// MODE 3: f32 out = residual*(acc+bias) + (1-residual)*aux.
// ---------------------------------------------------------------------------
template <int MODE>
__global__ __launch_bounds__(128)
void gemm_wmma_kernel(const _Float16* __restrict__ A,
                      const _Float16* __restrict__ Wt,   // [Nout][K] f16
                      const float* __restrict__ bias,
                      void* __restrict__ outv,
                      const float* __restrict__ aux,
                      const float* __restrict__ resPtr,
                      int M, int K, int Nout) {
  __shared__ _Float16 As[2][64 * 40];    // 64 rows x 32 k, stride 40 halves
  __shared__ _Float16 Bts[2][64 * 40];   // 64 cols x 32 k, stride 40 halves

  const int t    = threadIdx.x;
  const int lane = t & 31;
  const int wave = t >> 5;
  const int wm   = wave >> 1;            // 0..1 : 32-row strip
  const int wn   = wave & 1;             // 0..1 : 32-col strip
  const int m0   = blockIdx.x * 64;
  const int n0   = blockIdx.y * 64;

  v8f acc[2][2] = {};
  const int nk = K / 32;

#if USE_TDM
  auto issue = [&](int ki, int buf) {
    const int k0 = ki * 32;
    tdm_load_2d((unsigned)(uintptr_t)&As[buf][0],
                A + (size_t)m0 * K + k0, (unsigned)K, (unsigned)M,
                32u, 64u, (unsigned)K);
    tdm_load_2d((unsigned)(uintptr_t)&Bts[buf][0],
                Wt + (size_t)n0 * K + k0, (unsigned)K, (unsigned)Nout,
                32u, 64u, (unsigned)K);
  };
  if (wave == 0) issue(0, 0);
#endif

  for (int ki = 0; ki < nk; ++ki) {
    const int cur = ki & 1;
#if USE_TDM
    if (wave == 0) {
      if (ki + 1 < nk) {
        issue(ki + 1, cur ^ 1);              // prefetch next tile pair
        __builtin_amdgcn_s_wait_tensorcnt(2); // two oldest (current) done
      } else {
        __builtin_amdgcn_s_wait_tensorcnt(0);
      }
    }
    __syncthreads();
#else
    // Fallback: cooperative loads (A and Wt are both K-contiguous).
    const int k0 = ki * 32;
    #pragma unroll
    for (int i = 0; i < 2; ++i) {
      int c  = t + i * 128;
      int r  = c >> 2;
      int ko = (c & 3) * 8;
      *(v8h*)&As[cur][r * 40 + ko] =
          *(const v8h*)&A[(size_t)(m0 + r) * K + k0 + ko];
      *(v8h*)&Bts[cur][r * 40 + ko] =
          *(const v8h*)&Wt[(size_t)(n0 + r) * K + k0 + ko];
    }
    __syncthreads();
#endif

    v16h af[2], bf[2];
    #pragma unroll
    for (int rs = 0; rs < 2; ++rs)
      af[rs] = load_a_frag(&As[cur][(wm * 32 + rs * 16) * 40], 40);
    #pragma unroll
    for (int cs = 0; cs < 2; ++cs)
      bf[cs] = load_bt_frag(&Bts[cur][(wn * 32 + cs * 16) * 40], 40);
    #pragma unroll
    for (int rs = 0; rs < 2; ++rs)
      #pragma unroll
      for (int cs = 0; cs < 2; ++cs)
        acc[rs][cs] = wmma32(af[rs], bf[cs], acc[rs][cs]);
    __syncthreads();
  }

  const float residual = (MODE == 3) ? *resPtr : 0.f;
  const int rbase = (lane >> 4) * 8;   // C layout: VGPR i -> row i or i+8
  const int col0  = lane & 15;
  #pragma unroll
  for (int rs = 0; rs < 2; ++rs) {
    #pragma unroll
    for (int cs = 0; cs < 2; ++cs) {
      const int gcol = n0 + wn * 32 + cs * 16 + col0;
      const float bcol = bias[gcol];
      #pragma unroll
      for (int i = 0; i < 8; ++i) {
        const int grow = m0 + wm * 32 + rs * 16 + rbase + i;
        const size_t idx = (size_t)grow * Nout + gcol;
        float v = acc[rs][cs][i] + bcol;
        if constexpr (MODE == 0) {
          ((_Float16*)outv)[idx] = (_Float16)v;
        } else if constexpr (MODE == 1) {
          float e = v > 0.f ? v : (__expf(v) - 1.f);   // ELU (alpha=1)
          ((_Float16*)outv)[idx] = (_Float16)e;
        } else if constexpr (MODE == 2) {
          ((float*)outv)[idx] = v;
        } else {
          ((float*)outv)[idx] = residual * v + (1.f - residual) * aux[idx];
        }
      }
    }
  }
}

// ---------------------------------------------------------------------------
// Flash-style attention. One wave per (16-row n-tile, 128-col d-chunk, batch).
// scores = (Q K^T / sqrt(D)) * exp(-(adj-scale)^2/width); softmax over m; @V.
// ---------------------------------------------------------------------------
__global__ __launch_bounds__(32)
void attn_wmma_kernel(const _Float16* __restrict__ Qh,
                      const _Float16* __restrict__ Kh,
                      const _Float16* __restrict__ Vt,   // [B][D][N]
                      const float* __restrict__ adj,
                      const float* __restrict__ scaleP,
                      const float* __restrict__ widthP,
                      _Float16* __restrict__ Hout) {
  const int lane = threadIdx.x & 31;
  const int n0 = blockIdx.x * 16;
  const int d0 = blockIdx.y * 128;
  const int b  = blockIdx.z;

  const _Float16* Qb  = Qh + (size_t)b * Nn * Dd;
  const _Float16* Kb  = Kh + (size_t)b * Nn * Dd;
  const _Float16* Vb  = Vt + (size_t)b * Dd * Nn;
  const float*    adjb = adj + (size_t)b * Nn * Nn;
  const float scale = *scaleP;
  const float invw  = 1.f / *widthP;
  const float invsq = 0.044194173824159216f;  // 1/sqrt(512)

  __shared__ _Float16 Ptile[16 * 40];   // C-layout -> A-layout staging

  v8f hacc[8] = {};
  float mrun[8], lrun[8];
  #pragma unroll
  for (int i = 0; i < 8; ++i) { mrun[i] = -1e30f; lrun[i] = 0.f; }

  const int rbase = (lane >> 4) * 8;
  const int col0  = lane & 15;

  for (int m0 = 0; m0 < Nn; m0 += 32) {
    if (m0 + 32 < Nn)
      __builtin_prefetch(adjb + (size_t)(n0 + rbase) * Nn + m0 + 32 + col0, 0, 0);

    // ---- score tiles S0 (cols m0..m0+15), S1 (m0+16..m0+31) ----
    v8f s0 = {}, s1 = {};
    for (int d = 0; d < Dd; d += 32) {
      v16h aq  = load_a_frag(Qb + (size_t)n0 * Dd + d, Dd);
      v16h bk0 = load_bt_frag(Kb + (size_t)m0 * Dd + d, Dd);
      v16h bk1 = load_bt_frag(Kb + (size_t)(m0 + 16) * Dd + d, Dd);
      s0 = wmma32(aq, bk0, s0);
      s1 = wmma32(aq, bk1, s1);
    }
    // ---- mask, scale, online softmax (rows live in 16-lane groups) ----
    float mnew[8];
    #pragma unroll
    for (int i = 0; i < 8; ++i) {
      const int grow = n0 + rbase + i;
      const float a0 = adjb[(size_t)grow * Nn + m0 + col0];
      const float a1 = adjb[(size_t)grow * Nn + m0 + 16 + col0];
      const float k0 = __expf(-(a0 - scale) * (a0 - scale) * invw);
      const float k1 = __expf(-(a1 - scale) * (a1 - scale) * invw);
      s0[i] = s0[i] * invsq * k0;
      s1[i] = s1[i] * invsq * k1;
      float v = fmaxf(s0[i], s1[i]);
      v = fmaxf(v, __shfl_xor(v, 1, 16));
      v = fmaxf(v, __shfl_xor(v, 2, 16));
      v = fmaxf(v, __shfl_xor(v, 4, 16));
      v = fmaxf(v, __shfl_xor(v, 8, 16));
      mnew[i] = fmaxf(mrun[i], v);
    }
    #pragma unroll
    for (int i = 0; i < 8; ++i) {
      const float corr = __expf(mrun[i] - mnew[i]);
      lrun[i] *= corr;
      #pragma unroll
      for (int j = 0; j < 8; ++j) hacc[j][i] *= corr;
      const float p0 = __expf(s0[i] - mnew[i]);
      const float p1 = __expf(s1[i] - mnew[i]);
      float ps = p0 + p1;
      ps += __shfl_xor(ps, 1, 16);
      ps += __shfl_xor(ps, 2, 16);
      ps += __shfl_xor(ps, 4, 16);
      ps += __shfl_xor(ps, 8, 16);
      lrun[i] += ps;
      mrun[i] = mnew[i];
      const int prow = rbase + i;
      Ptile[prow * 40 + col0]      = (_Float16)p0;
      Ptile[prow * 40 + 16 + col0] = (_Float16)p1;
    }
    __syncthreads();
    // ---- P (16x32) @ V (32x128 via Vt rows) ----
    v16h ap = load_a_frag(Ptile, 40);
    #pragma unroll
    for (int j = 0; j < 8; ++j) {
      v16h bv = load_bt_frag(Vb + (size_t)(d0 + j * 16) * Nn + m0, Nn);
      hacc[j] = wmma32(ap, bv, hacc[j]);
    }
    __syncthreads();
  }
  // ---- normalize and store h (f16) ----
  #pragma unroll
  for (int j = 0; j < 8; ++j) {
    const int gd = d0 + j * 16 + col0;
    #pragma unroll
    for (int i = 0; i < 8; ++i) {
      const int grow = n0 + rbase + i;
      const float o = hacc[j][i] / lrun[i];
      Hout[((size_t)b * Nn + grow) * Dd + gd] = (_Float16)o;
    }
  }
}

// ---------------------------------------------------------------------------
// Host-side orchestration
// ---------------------------------------------------------------------------
extern "C" void kernel_launch(void* const* d_in, const int* in_sizes, int n_in,
                              void* d_out, int out_size, void* d_ws, size_t ws_size,
                              hipStream_t stream) {
  (void)in_sizes; (void)n_in; (void)out_size; (void)ws_size;

  const float* x    = (const float*)d_in[0];
  const float* adj  = (const float*)d_in[1];
  const float* Wq   = (const float*)d_in[2];
  const float* bq   = (const float*)d_in[3];
  const float* Wk   = (const float*)d_in[4];
  const float* bk   = (const float*)d_in[5];
  const float* Wv   = (const float*)d_in[6];
  const float* bv   = (const float*)d_in[7];
  const float* W1   = (const float*)d_in[8];
  const float* b1   = (const float*)d_in[9];
  const float* W2   = (const float*)d_in[10];
  const float* b2   = (const float*)d_in[11];
  const float* Wp   = (const float*)d_in[12];
  const float* bp   = (const float*)d_in[13];
  const float* scaleP = (const float*)d_in[14];
  const float* widthP = (const float*)d_in[15];
  const float* resP   = (const float*)d_in[16];

  char* ws = (char*)d_ws;
  size_t off = 0;
  auto alloc = [&](size_t bytes) -> char* {
    char* p = ws + off;
    off += (bytes + 255) & ~(size_t)255;
    return p;
  };
  const size_t xBytesH = (size_t)Mrows * Dd * sizeof(_Float16);
  const size_t wBytesH = (size_t)Dd * Dd * sizeof(_Float16);

  _Float16* xh   = (_Float16*)alloc(xBytesH);
  _Float16* Wqt  = (_Float16*)alloc(wBytesH);   // transposed f16 weights [N][K]
  _Float16* Wkt  = (_Float16*)alloc(wBytesH);
  _Float16* Wvt  = (_Float16*)alloc(wBytesH);
  _Float16* W1t  = (_Float16*)alloc(wBytesH);
  _Float16* W2t  = (_Float16*)alloc(wBytesH);
  _Float16* Wpt  = (_Float16*)alloc(wBytesH);
  _Float16* Qh   = (_Float16*)alloc(xBytesH);
  _Float16* Kh   = (_Float16*)alloc(xBytesH);
  _Float16* Vh   = (_Float16*)alloc(xBytesH);
  _Float16* Vt   = (_Float16*)alloc(xBytesH);
  _Float16* hTmp = (_Float16*)alloc(xBytesH);
  _Float16* tTmp = (_Float16*)alloc(xBytesH);
  float*    Pxp  = (float*)   alloc((size_t)Mrows * Dd * sizeof(float));

  // 1) precision converts (+ weight transpose so GEMM B tiles are K-contiguous)
  {
    const int nX = Mrows * Dd;
    f32_to_f16_kernel<<<8192, 256, 0, stream>>>(x, xh, nX);
    dim3 gW(Dd / 32, Dd / 32);   // (16, 16)
    convert_transpose_w_kernel<<<gW, 256, 0, stream>>>(Wq, Wqt);
    convert_transpose_w_kernel<<<gW, 256, 0, stream>>>(Wk, Wkt);
    convert_transpose_w_kernel<<<gW, 256, 0, stream>>>(Wv, Wvt);
    convert_transpose_w_kernel<<<gW, 256, 0, stream>>>(W1, W1t);
    convert_transpose_w_kernel<<<gW, 256, 0, stream>>>(W2, W2t);
    convert_transpose_w_kernel<<<gW, 256, 0, stream>>>(Wp, Wpt);
  }

  const dim3 gGemm(Mrows / 64, Dd / 64);   // (256, 8)
  const dim3 bGemm(128);

  // 2) projections
  gemm_wmma_kernel<0><<<gGemm, bGemm, 0, stream>>>(xh, Wqt, bq, Qh, nullptr, nullptr, Mrows, Dd, Dd);
  gemm_wmma_kernel<0><<<gGemm, bGemm, 0, stream>>>(xh, Wkt, bk, Kh, nullptr, nullptr, Mrows, Dd, Dd);
  gemm_wmma_kernel<0><<<gGemm, bGemm, 0, stream>>>(xh, Wvt, bv, Vh, nullptr, nullptr, Mrows, Dd, Dd);
  gemm_wmma_kernel<2><<<gGemm, bGemm, 0, stream>>>(xh, Wpt, bp, Pxp, nullptr, nullptr, Mrows, Dd, Dd);

  // 3) V transpose for PV B-fragments
  {
    dim3 g(Nn / 32, Dd / 32, Bb);   // (64, 16, 8)
    transpose_f16_kernel<<<g, 256, 0, stream>>>(Vh, Vt);
  }

  // 4) attention
  {
    dim3 g(Nn / 16, Dd / 128, Bb);  // (128, 4, 8)
    attn_wmma_kernel<<<g, 32, 0, stream>>>(Qh, Kh, Vt, adj, scaleP, widthP, hTmp);
  }

  // 5) FFN + residual blend
  gemm_wmma_kernel<1><<<gGemm, bGemm, 0, stream>>>(hTmp, W1t, b1, tTmp, nullptr, nullptr, Mrows, Dd, Dd);
  gemm_wmma_kernel<3><<<gGemm, bGemm, 0, stream>>>(tTmp, W2t, b2, (float*)d_out, Pxp, resP, Mrows, Dd, Dd);
}
